// TernaryFusedMoEMethod_32040456028412
// MI455X (gfx1250) — compile-verified, matrix operand order checked by reading, not run on hardware
//
#include <hip/hip_runtime.h>
#include <hip/hip_bf16.h>
#include <math.h>

typedef __attribute__((ext_vector_type(16))) _Float16 v16h;
typedef __attribute__((ext_vector_type(8)))  _Float16 v8h;
typedef __attribute__((ext_vector_type(8)))  float    v8f;

#define T_TOK   4096
#define H_DIM   2048
#define I_DIM   768
#define NEXP    16
#define TILE_T  32
#define MAX_TILES (T_TOK / TILE_T)   // worst case: one expert receives every token

// LDS strides chosen for 4-bank row rotation (conflict-free 16-row access)
// and 16B row alignment (ds_load_b128).
#define XS_STRIDE  2056              // 2048 + 8  ; 4112 B per row
#define ACT_STRIDE 776               // 768  + 8  ; 1552 B per row

// LDS carve order (low offsets first so DS offset immediates can fold):
//   lut (2 KB) | tok (128 B) | w (128 B) | act (48.5 KB) | raw (16 KB) | xs (128.5 KB)
#define LUT_BYTES  (256 * 8)                          //   2048
#define MISC_BYTES ((size_t)TILE_T * 8)               //    256 (tok + w)
#define ACT_BYTES  ((size_t)TILE_T * ACT_STRIDE * 2)  //  49664
#define RAW_BYTES  ((size_t)2 * TILE_T * 64 * 4)      //  16384 (double buffer)
#define XS_BYTES   ((size_t)TILE_T * XS_STRIDE * 2)   // 131584
#define OFF_TOK    (LUT_BYTES)
#define OFF_W      (OFF_TOK + (size_t)TILE_T * 4)
#define OFF_ACT    (LUT_BYTES + MISC_BYTES)
#define OFF_RAW    (OFF_ACT + ACT_BYTES)
#define OFF_XS     (OFF_RAW + RAW_BYTES)
#define SHMEM_BYTES (OFF_XS + XS_BYTES)

#if __has_builtin(__builtin_amdgcn_global_load_async_to_lds_b128)
#define HAVE_ASYNC_LDS 1
typedef int async_v4i __attribute__((vector_size(16)));
#endif

// ---------------------------------------------------------------------------
// helpers
// ---------------------------------------------------------------------------
__device__ __forceinline__ v8f wmma_f16(v16h a, v16h b, v8f c) {
  // (neg_a, A, neg_b, B, c_mod, C, reuse_a, reuse_b)
  return __builtin_amdgcn_wmma_f32_16x16x32_f16(false, a, false, b,
                                                (short)0, c, false, false);
}

// A-strip per documented 16-bit 16x32 layout: lane holds row (lane%16),
// K = {k0..k0+7} U {k0+16..k0+23}, where k0 = ks + (lane>>4)*8.
__device__ __forceinline__ v16h load_a_strip(const _Float16* sh, int stride,
                                             int row, int k0) {
  v8h lo = *(const v8h*)(sh + row * stride + k0);
  v8h hi = *(const v8h*)(sh + row * stride + k0 + 16);
  v16h r;
#pragma unroll
  for (int j = 0; j < 8; ++j) { r[j] = lo[j]; r[8 + j] = hi[j]; }
  return r;
}

// LUT-based ternary unpack: one byte (4 codes) -> one u64 (4 f16) via ds_load_b64.
// LUT lives at LDS offset 0, so address is just (byte << 3).
__device__ __forceinline__ v16h lut_unpack(const unsigned long long* lut,
                                           unsigned int lo16, unsigned int hi16) {
  union { v16h v; unsigned long long q[4]; } u;
  u.q[0] = lut[lo16 & 0xFFu];
  u.q[1] = lut[(lo16 >> 8) & 0xFFu];
  u.q[2] = lut[hi16 & 0xFFu];
  u.q[3] = lut[(hi16 >> 8) & 0xFFu];
  return u.v;
}

// ---------------------------------------------------------------------------
// kernel 1: zero output + counters
// ---------------------------------------------------------------------------
__global__ void moe_init_kernel(float* __restrict__ out, int* __restrict__ counts,
                                long long n) {
  long long i = (long long)blockIdx.x * blockDim.x + threadIdx.x;
  if (i < n) out[i] = 0.0f;
  if (i < NEXP) counts[i] = 0;
}

// ---------------------------------------------------------------------------
// kernel 2: softmax top-2 routing, bucket tokens per expert
// ---------------------------------------------------------------------------
__global__ void moe_route_kernel(const float* __restrict__ logits,
                                 int* __restrict__ counts,
                                 int* __restrict__ lists,
                                 float* __restrict__ listw) {
  int t = blockIdx.x * blockDim.x + threadIdx.x;
  if (t >= T_TOK) return;
  float l[NEXP];
  float mx = -3.0e38f;
#pragma unroll
  for (int e = 0; e < NEXP; ++e) { l[e] = logits[t * NEXP + e]; mx = fmaxf(mx, l[e]); }
  float p[NEXP];
#pragma unroll
  for (int e = 0; e < NEXP; ++e) p[e] = __expf(l[e] - mx);
  int i1 = 0;
#pragma unroll
  for (int e = 1; e < NEXP; ++e) if (p[e] > p[i1]) i1 = e;
  int i2 = (i1 == 0) ? 1 : 0;
#pragma unroll
  for (int e = 0; e < NEXP; ++e) if (e != i1 && p[e] > p[i2]) i2 = e;
  float w1 = p[i1] / (p[i1] + p[i2]);
  float w2 = 1.0f - w1;
  int s1 = atomicAdd(&counts[i1], 1);
  lists[i1 * T_TOK + s1] = t;  listw[i1 * T_TOK + s1] = w1;
  int s2 = atomicAdd(&counts[i2], 1);
  lists[i2 * T_TOK + s2] = t;  listw[i2 * T_TOK + s2] = w2;
}

// ---------------------------------------------------------------------------
// kernel 3: fused expert FFN for a 32-token tile of one expert
//   stage:   xs[32][2048] = (x * a13) as f16, async-copied + converted, once
//   phase 1: act = silu(xs @ Wg^T) * (xs @ Wu^T), scaled by a2   (barrier-free)
//   phase 2: out += route_w * (act_s @ W2^T)                     (barrier-free)
// ---------------------------------------------------------------------------
__global__ __launch_bounds__(256) void moe_expert_kernel(
    const float* __restrict__ x,
    const unsigned char* __restrict__ w13p,
    const signed char*  __restrict__ a13q, const float* __restrict__ a13s_p,
    const unsigned char* __restrict__ w2p,
    const signed char*  __restrict__ a2q,  const float* __restrict__ a2s_p,
    const int* __restrict__ counts, const int* __restrict__ lists,
    const float* __restrict__ listw,
    float* __restrict__ out) {
  extern __shared__ __align__(16) char smem[];
  unsigned long long* sh_lut  = (unsigned long long*)smem;       // offset 0: cheap DS addr
  int*                sh_tok  = (int*)(smem + OFF_TOK);
  float*              sh_w    = (float*)(smem + OFF_W);
  _Float16*           sh_act  = (_Float16*)(smem + OFF_ACT);     // below 64 KB
  float*              sh_xraw = (float*)(smem + OFF_RAW);
  _Float16*           sh_x    = (_Float16*)(smem + OFF_XS);      // big tile last

  const int e     = blockIdx.y;
  const int cnt   = counts[e];
  const int tile0 = blockIdx.x * TILE_T;
  if (tile0 >= cnt) return;

  const int tid   = threadIdx.x;
  const int wave  = tid >> 5;
  const int lane  = tid & 31;
  const int nloc  = lane & 15;   // matrix row (A) / column (B,D) within 16-tile
  const int khalf = lane >> 4;   // which K interleave half

  const float a13s = a13s_p[0];
  const float a2s  = a2s_p[0];

  // LUT: byte (4 ternary codes {0,1,2}) -> 4 packed f16 in {-1,0,+1}
  {
    unsigned int b = tid & 255;
    unsigned long long q = 0;
#pragma unroll
    for (int i = 0; i < 4; ++i) {
      unsigned int c = (b >> (2 * i)) & 3u;
      unsigned long long h = (c == 1u) ? 0x0000ull : ((c == 0u) ? 0xBC00ull : 0x3C00ull);
      q |= h << (16 * i);
    }
    sh_lut[tid & 255] = q;
  }
  if (tid < TILE_T) {
    int slot = tile0 + tid;
    int sc   = slot < cnt ? slot : (cnt - 1);
    sh_tok[tid] = lists[e * T_TOK + sc];
    sh_w[tid]   = (slot < cnt) ? listw[e * T_TOK + sc] : 0.0f;
  }
  __syncthreads();

  // ---------------- stage: xs = (x * a13) as f16, single pass ---------------
#ifdef HAVE_ASYNC_LDS
  // prologue: async chunk 0 into raw buffer 0 (512 x 16B segments per chunk)
#pragma unroll
  for (int q = 0; q < 2; ++q) {
    int s = q * 256 + tid;
    int r = s >> 4;
    int c = (s & 15) * 4;
    __builtin_amdgcn_global_load_async_to_lds_b128(
        (__attribute__((address_space(1))) async_v4i*)(void*)(x + (long long)sh_tok[r] * H_DIM + c),
        (__attribute__((address_space(3))) async_v4i*)(void*)&sh_xraw[r * 64 + c],
        0, 0);
  }
#endif
  for (int kc = 0; kc < H_DIM; kc += 64) {
#ifdef HAVE_ASYNC_LDS
    const int buf = (kc >> 6) & 1;
#if __has_builtin(__builtin_amdgcn_s_wait_asynccnt)
    __builtin_amdgcn_s_wait_asynccnt(0);
#else
    asm volatile("s_wait_asynccnt 0" ::: "memory");
#endif
    __syncthreads();                       // raw[buf] complete for all waves
    if (kc + 64 < H_DIM) {                 // overlap: issue next chunk now
#pragma unroll
      for (int q = 0; q < 2; ++q) {
        int s = q * 256 + tid;
        int r = s >> 4;
        int c = (s & 15) * 4;
        __builtin_amdgcn_global_load_async_to_lds_b128(
            (__attribute__((address_space(1))) async_v4i*)(void*)(x + (long long)sh_tok[r] * H_DIM + kc + 64 + c),
            (__attribute__((address_space(3))) async_v4i*)(void*)&sh_xraw[(buf ^ 1) * TILE_T * 64 + r * 64 + c],
            0, 0);
      }
    }
#endif
    // scale by a13, convert to f16, place into the full xs tile
#pragma unroll
    for (int q = 0; q < 8; ++q) {
      int idx = q * 256 + tid;             // 0..2047
      int r   = idx >> 6;
      int c   = idx & 63;
#ifdef HAVE_ASYNC_LDS
      float xv = sh_xraw[buf * TILE_T * 64 + r * 64 + c];
#else
      float xv = x[(long long)sh_tok[r] * H_DIM + kc + c];
#endif
      float av = a13s * (float)a13q[e * H_DIM + kc + c];
      sh_x[r * XS_STRIDE + kc + c] = (_Float16)(xv * av);
    }
  }
  __syncthreads();

  // ---------------- phase 1: gate/up GEMM + silu (no barriers) --------------
  const long long w13_base = (long long)e * (2 * I_DIM) * (H_DIM / 4);

  for (int pp = 0; pp < 3; ++pp) {
    const int it0   = wave * 6 + pp * 2;     // pair of 16-wide i-tiles
    const int colg0 = it0 * 16 + nloc;       // gate output columns
    const int colg1 = colg0 + 16;
    const unsigned char* wg0 = w13p + w13_base + (long long)colg0 * (H_DIM / 4);
    const unsigned char* wu0 = w13p + w13_base + (long long)(colg0 + I_DIM) * (H_DIM / 4);
    const unsigned char* wg1 = w13p + w13_base + (long long)colg1 * (H_DIM / 4);
    const unsigned char* wu1 = w13p + w13_base + (long long)(colg1 + I_DIM) * (H_DIM / 4);
    __builtin_prefetch(wg0, 0, 0); __builtin_prefetch(wu0, 0, 0);
    __builtin_prefetch(wg1, 0, 0); __builtin_prefetch(wu1, 0, 0);

    v8f g0a = {}; v8f g0b = {}; v8f u0a = {}; v8f u0b = {};
    v8f g1a = {}; v8f g1b = {}; v8f u1a = {}; v8f u1b = {};

#pragma unroll 2
    for (int ks = 0; ks < H_DIM; ks += 32) {
      const int k0 = ks + khalf * 8;         // this lane's K base
      v16h a0 = load_a_strip(sh_x, XS_STRIDE, nloc,      k0);
      v16h a1 = load_a_strip(sh_x, XS_STRIDE, 16 + nloc, k0);

      unsigned int g0lo = *(const unsigned short*)(wg0 + (k0 >> 2));
      unsigned int g0hi = *(const unsigned short*)(wg0 + ((k0 + 16) >> 2));
      unsigned int u0lo = *(const unsigned short*)(wu0 + (k0 >> 2));
      unsigned int u0hi = *(const unsigned short*)(wu0 + ((k0 + 16) >> 2));
      unsigned int g1lo = *(const unsigned short*)(wg1 + (k0 >> 2));
      unsigned int g1hi = *(const unsigned short*)(wg1 + ((k0 + 16) >> 2));
      unsigned int u1lo = *(const unsigned short*)(wu1 + (k0 >> 2));
      unsigned int u1hi = *(const unsigned short*)(wu1 + ((k0 + 16) >> 2));

      v16h bg0 = lut_unpack(sh_lut, g0lo, g0hi);
      v16h bu0 = lut_unpack(sh_lut, u0lo, u0hi);
      v16h bg1 = lut_unpack(sh_lut, g1lo, g1hi);
      v16h bu1 = lut_unpack(sh_lut, u1lo, u1hi);

      g0a = wmma_f16(a0, bg0, g0a);  g0b = wmma_f16(a1, bg0, g0b);
      u0a = wmma_f16(a0, bu0, u0a);  u0b = wmma_f16(a1, bu0, u0b);
      g1a = wmma_f16(a0, bg1, g1a);  g1b = wmma_f16(a1, bg1, g1b);
      u1a = wmma_f16(a0, bu1, u1a);  u1b = wmma_f16(a1, bu1, u1b);
    }

    // epilogue: silu(g)*u, fold in a2 (per i column), store f16 to LDS
    const float a2v0 = a2s * (float)a2q[e * I_DIM + colg0];
    const float a2v1 = a2s * (float)a2q[e * I_DIM + colg1];
#pragma unroll
    for (int j = 0; j < 8; ++j) {
      int row = j + khalf * 8;               // documented f32 C/D layout
      float g, u, s;
      g = g0a[j]; u = u0a[j]; s = (g / (1.0f + __expf(-g))) * u;
      sh_act[row * ACT_STRIDE + colg0] = (_Float16)(s * a2v0);
      g = g0b[j]; u = u0b[j]; s = (g / (1.0f + __expf(-g))) * u;
      sh_act[(16 + row) * ACT_STRIDE + colg0] = (_Float16)(s * a2v0);
      g = g1a[j]; u = u1a[j]; s = (g / (1.0f + __expf(-g))) * u;
      sh_act[row * ACT_STRIDE + colg1] = (_Float16)(s * a2v1);
      g = g1b[j]; u = u1b[j]; s = (g / (1.0f + __expf(-g))) * u;
      sh_act[(16 + row) * ACT_STRIDE + colg1] = (_Float16)(s * a2v1);
    }
  }
  __syncthreads();

  // ---------------- phase 2: down GEMM + weighted scatter --------------------
  const long long w2_base = (long long)e * H_DIM * (I_DIM / 4);

  for (int htb = 0; htb < 16; ++htb) {
    const int h_tile = wave * 16 + htb;      // 0..127 (16-wide tiles over H)
    const int h      = h_tile * 16 + nloc;
    const unsigned char* wr = w2p + w2_base + (long long)h * (I_DIM / 4);
    if (htb < 15) __builtin_prefetch(wr + 16 * (I_DIM / 4), 0, 0);

    v8f acc0 = {}; v8f acc1 = {};
#pragma unroll 2
    for (int ks = 0; ks < I_DIM; ks += 32) {
      const int k0 = ks + khalf * 8;
      v16h a0 = load_a_strip(sh_act, ACT_STRIDE, nloc,      k0);
      v16h a1 = load_a_strip(sh_act, ACT_STRIDE, 16 + nloc, k0);
      unsigned int lo = *(const unsigned short*)(wr + (k0 >> 2));
      unsigned int hi = *(const unsigned short*)(wr + ((k0 + 16) >> 2));
      v16h b = lut_unpack(sh_lut, lo, hi);
      acc0 = wmma_f16(a0, b, acc0);
      acc1 = wmma_f16(a1, b, acc1);
    }
#pragma unroll
    for (int j = 0; j < 8; ++j) {
      int row = j + khalf * 8;
      int t0 = sh_tok[row];
      atomicAdd(&out[(long long)t0 * H_DIM + h], sh_w[row] * acc0[j]);
      int t1 = sh_tok[16 + row];
      atomicAdd(&out[(long long)t1 * H_DIM + h], sh_w[16 + row] * acc1[j]);
    }
  }
}

// ---------------------------------------------------------------------------
// launch
// ---------------------------------------------------------------------------
extern "C" void kernel_launch(void* const* d_in, const int* in_sizes, int n_in,
                              void* d_out, int out_size, void* d_ws, size_t ws_size,
                              hipStream_t stream) {
  (void)in_sizes; (void)n_in; (void)ws_size;
  const float*         x      = (const float*)d_in[0];
  const float*         logits = (const float*)d_in[1];
  const unsigned char* w13p   = (const unsigned char*)d_in[2];
  const signed char*   a13q   = (const signed char*)d_in[3];
  const float*         a13s   = (const float*)d_in[4];
  const unsigned char* w2p    = (const unsigned char*)d_in[5];
  const signed char*   a2q    = (const signed char*)d_in[6];
  const float*         a2s    = (const float*)d_in[7];
  float*               out    = (float*)d_out;

  // workspace layout: counts[16] | lists[E*T] | listw[E*T]  (~525 KB)
  int*   counts = (int*)d_ws;
  int*   lists  = (int*)((char*)d_ws + 256);
  float* listw  = (float*)((char*)d_ws + 256 + (size_t)NEXP * T_TOK * sizeof(int));

  long long n = (long long)out_size;
  moe_init_kernel<<<(unsigned)((n + 255) / 256), 256, 0, stream>>>(out, counts, n);
  moe_route_kernel<<<(T_TOK + 127) / 128, 128, 0, stream>>>(logits, counts, lists, listw);

  dim3 grid(MAX_TILES, NEXP);
  moe_expert_kernel<<<grid, 256, (unsigned)SHMEM_BYTES, stream>>>(
      x, w13p, a13q, a13s, w2p, a2q, a2s, counts, lists, listw, out);
}
